// GATBackbone_13400297963689
// MI455X (gfx1250) — matrix operand order repaired and checked by reading, not compiled, not run on hardware
//
#include <hip/hip_runtime.h>

typedef __attribute__((ext_vector_type(16))) _Float16 v16h;
typedef __attribute__((ext_vector_type(8)))  _Float16 v8h;
typedef __attribute__((ext_vector_type(8)))  float    v8f;

#define NEG_SLOPE 0.2f
#define BN_EPS 1e-5f

// ---------------------------------------------------------------------------
// float atomic max via CAS (monotonic, header-portable)
// ---------------------------------------------------------------------------
__device__ __forceinline__ void atomicMaxF(float* addr, float val) {
  unsigned int* ua = (unsigned int*)addr;
  unsigned int old = *ua;
  while (__uint_as_float(old) < val) {
    unsigned int assumed = old;
    old = atomicCAS(ua, assumed, __float_as_uint(val));
    if (old == assumed) break;
  }
}

// ---------------------------------------------------------------------------
// generic fill
// ---------------------------------------------------------------------------
__global__ void fill_kernel(float* __restrict__ p, int n, float v) {
  int i = blockIdx.x * blockDim.x + threadIdx.x;
  if (i < n) p[i] = v;
}

// ---------------------------------------------------------------------------
// f32 [n,K] -> f16 [npad,Kpad] row-major, zero padded
// ---------------------------------------------------------------------------
__global__ void cvt_x(const float* __restrict__ X, _Float16* __restrict__ Xh,
                      int n, int K, int Kpad, int npad) {
  int i = blockIdx.x * blockDim.x + threadIdx.x;
  if (i >= npad * Kpad) return;
  int r = i / Kpad, k = i - r * Kpad;
  Xh[i] = (r < n && k < K) ? (_Float16)X[(size_t)r * K + k] : (_Float16)0.f;
}

// f32 W[K,ncols] -> f16 W^T[cpad,Kpad] row-major, zero padded
__global__ void cvt_wt(const float* __restrict__ W, _Float16* __restrict__ WT,
                       int K, int ncols, int Kpad, int cpad) {
  int i = blockIdx.x * blockDim.x + threadIdx.x;
  if (i >= cpad * Kpad) return;
  int c = i / Kpad, k = i - c * Kpad;
  WT[i] = (k < K && c < ncols) ? (_Float16)W[(size_t)k * ncols + c] : (_Float16)0.f;
}

// ---------------------------------------------------------------------------
// WMMA GEMM: Z[n,ldz] = Xh[Npad,Kpad] * WT[cpad,Kpad]^T, f32 accumulate.
// One wave32 per 16-row x (16*NT)-col strip; fragments loaded straight from
// global memory as aligned b128 vectors in contiguous k order (k = 16*hi + s, same for A and B):
//   A (16x32 f16): lane -> row = lane%16; halves 0-7 at K = kk+(lane<16?0:16),
//                  halves 8-15 at +8  (two v8h loads)
//   B (32x16 f16) from W^T: lane -> col = lane%16; halves 0-15 contiguous at
//                  K = kk+(lane<16?0:16)  (one aligned v16h load)
//   C/D (f32):     VGPR r -> M = r + (lane>=16?8:0), N = lane%16
// ---------------------------------------------------------------------------
template <int NT>
__global__ void gemm_wmma(const _Float16* __restrict__ Xh, const _Float16* __restrict__ WT,
                          float* __restrict__ Z, int n, int Kpad, int ldz) {
  const int lane = threadIdx.x;            // 32 = one wave
  const int row0 = blockIdx.x * 16;
  const int col0 = blockIdx.y * (16 * NT);
  const int m  = lane & 15;
  const int hi = (lane >> 4) & 1;

  v8f acc[NT];
#pragma unroll
  for (int t = 0; t < NT; ++t) acc[t] = (v8f){};

  const _Float16* xrow = Xh + (size_t)(row0 + m) * Kpad + hi * 16;
  for (int kk = 0; kk < Kpad; kk += 32) {
    v8h a0 = *(const v8h*)(xrow + kk);
    v8h a1 = *(const v8h*)(xrow + kk + 8);
    v16h af = __builtin_shufflevector(a0, a1, 0, 1, 2, 3, 4, 5, 6, 7,
                                      8, 9, 10, 11, 12, 13, 14, 15);
#pragma unroll
    for (int t = 0; t < NT; ++t) {
      const _Float16* wrow = WT + (size_t)(col0 + t * 16 + m) * Kpad + hi * 16 + kk;
      v16h bf = *(const v16h*)wrow;
      acc[t] = __builtin_amdgcn_wmma_f32_16x16x32_f16(
          /*neg_a=*/false, af, /*neg_b=*/false, bf,
          /*c_mod=*/(short)0, acc[t], /*reuse_a=*/false, /*reuse_b=*/false);
    }
  }

  const int mb = hi ? 8 : 0;
#pragma unroll
  for (int t = 0; t < NT; ++t)
#pragma unroll
    for (int r = 0; r < 8; ++r) {
      int grow = row0 + mb + r;
      if (grow < n) Z[(size_t)grow * ldz + col0 + t * 16 + m] = acc[t][r];
    }
}

// ---------------------------------------------------------------------------
// el[n,h] = sum_d z[n,h,d]*al[h,d] ; er likewise. One thread per (node, head).
// ---------------------------------------------------------------------------
__global__ void attn_coeff(const float* __restrict__ z, const float* __restrict__ al,
                           const float* __restrict__ ar, float* __restrict__ el,
                           float* __restrict__ er, int n, int outd, int ldz) {
  int idx = blockIdx.x * blockDim.x + threadIdx.x;
  if (idx >= n * 4) return;
  int node = idx >> 2, h = idx & 3;
  const float* zp = z + (size_t)node * ldz + h * outd;
  float sl = 0.f, sr = 0.f;
  for (int d = 0; d < outd; ++d) {
    float zv = zp[d];
    sl += zv * al[h * outd + d];
    sr += zv * ar[h * outd + d];
  }
  el[idx] = sl;
  er[idx] = sr;
}

// ---------------------------------------------------------------------------
// per-edge logits + leaky relu + segment max over dst. Self-loops appended.
// ---------------------------------------------------------------------------
__global__ void edge_logits_max(const float* __restrict__ el, const float* __restrict__ er,
                                const int* __restrict__ src, const int* __restrict__ dst,
                                int E, int n, float* __restrict__ ebuf,
                                float* __restrict__ mx) {
  int idx = blockIdx.x * blockDim.x + threadIdx.x;
  if (idx >= (E + n) * 4) return;
  int e = idx >> 2, h = idx & 3;
  int sN = (e < E) ? src[e] : (e - E);
  int dN = (e < E) ? dst[e] : (e - E);
  float v = el[sN * 4 + h] + er[dN * 4 + h];
  v = (v > 0.f) ? v : NEG_SLOPE * v;
  ebuf[idx] = v;
  atomicMaxF(&mx[dN * 4 + h], v);
}

// a = exp(e - max[dst]); segment-sum over dst
__global__ void edge_exp_sum(float* __restrict__ ebuf, const float* __restrict__ mx,
                             float* __restrict__ ss, const int* __restrict__ dst,
                             int E, int n) {
  int idx = blockIdx.x * blockDim.x + threadIdx.x;
  if (idx >= (E + n) * 4) return;
  int e = idx >> 2, h = idx & 3;
  int dN = (e < E) ? dst[e] : (e - E);
  float a = __expf(ebuf[idx] - mx[dN * 4 + h]);
  ebuf[idx] = a;
  atomicAdd(&ss[dN * 4 + h], a);
}

// out[dst,h,d] += (a / s[dst,h]) * z[src,h,d]   (OUTD compile-time: shifts)
template <int OUTD>
__global__ void edge_scatter(const float* __restrict__ ebuf, const float* __restrict__ ss,
                             const float* __restrict__ z, const int* __restrict__ src,
                             const int* __restrict__ dst, int E, int n,
                             float* __restrict__ out, int ldz) {
  long long idx = (long long)blockIdx.x * blockDim.x + threadIdx.x;
  long long total = (long long)(E + n) * 4 * OUTD;
  if (idx >= total) return;
  int d = (int)(idx % OUTD);
  long long t = idx / OUTD;
  int h = (int)(t & 3);
  int e = (int)(t >> 2);
  int sN = (e < E) ? src[e] : (e - E);
  int dN = (e < E) ? dst[e] : (e - E);
  float alpha = ebuf[e * 4 + h] / ss[dN * 4 + h];
  atomicAdd(&out[(size_t)dN * (4 * OUTD) + h * OUTD + d],
            alpha * z[(size_t)sN * ldz + h * OUTD + d]);
}

// ---------------------------------------------------------------------------
// head-sum + bias -> hout[n,64]; per-column sum / sum-of-squares via LDS
// atomics, one global atomic per column per block
// ---------------------------------------------------------------------------
__global__ void headsum_bn_stats(const float* __restrict__ oacc, const float* __restrict__ b,
                                 float* __restrict__ hout, float* __restrict__ colsum,
                                 float* __restrict__ colsq, int n) {
  __shared__ float s_sum[64];
  __shared__ float s_sq[64];
  int tid = threadIdx.x;  // 256
  if (tid < 64) { s_sum[tid] = 0.f; s_sq[tid] = 0.f; }
  __syncthreads();
  int idx = blockIdx.x * 256 + tid;
  int node = idx >> 6, d = idx & 63;
  if (node < n) {
    float v = 0.f;
#pragma unroll
    for (int h = 0; h < 4; ++h)
      v += oacc[(size_t)node * 256 + h * 64 + d] + b[h * 64 + d];
    hout[idx] = v;
    atomicAdd(&s_sum[d], v);
    atomicAdd(&s_sq[d], v * v);
  }
  __syncthreads();
  if (tid < 64) {
    atomicAdd(&colsum[tid], s_sum[tid]);
    atomicAdd(&colsq[tid], s_sq[tid]);
  }
}

__global__ void finalize_stats(const float* __restrict__ colsum, const float* __restrict__ colsq,
                               float* __restrict__ mu, float* __restrict__ rsig, int n) {
  int d = threadIdx.x;
  if (d < 64) {
    float m = colsum[d] / (float)n;
    float var = colsq[d] / (float)n - m * m;
    mu[d] = m;
    rsig[d] = rsqrtf(var + BN_EPS);
  }
}

__global__ void bn_apply(float* __restrict__ h, const float* __restrict__ mu,
                         const float* __restrict__ rsig, const float* __restrict__ gamma,
                         const float* __restrict__ beta, int n) {
  int i = blockIdx.x * blockDim.x + threadIdx.x;
  if (i >= n * 64) return;
  int d = i & 63;
  h[i] = (h[i] - mu[d]) * rsig[d] * gamma[d] + beta[d];
}

// final layer: d_out[n,c] = sum_h (oacc[n,h,c] + b2[h,c]), C = 2
__global__ void headsum_final(const float* __restrict__ oacc, const float* __restrict__ b,
                              float* __restrict__ out, int n) {
  int i = blockIdx.x * blockDim.x + threadIdx.x;
  if (i >= n * 2) return;
  int node = i >> 1, c = i & 1;
  float v = 0.f;
#pragma unroll
  for (int h = 0; h < 4; ++h) v += oacc[(size_t)node * 8 + h * 2 + c] + b[h * 2 + c];
  out[i] = v;
}

// ---------------------------------------------------------------------------
// host driver
// ---------------------------------------------------------------------------
extern "C" void kernel_launch(void* const* d_in, const int* in_sizes, int n_in,
                              void* d_out, int out_size, void* d_ws, size_t ws_size,
                              hipStream_t stream) {
  const float* feats = (const float*)d_in[0];
  const float* W1    = (const float*)d_in[1];
  const float* al1   = (const float*)d_in[2];
  const float* ar1   = (const float*)d_in[3];
  const float* b1    = (const float*)d_in[4];
  const float* Wm    = (const float*)d_in[5];
  const float* alm   = (const float*)d_in[6];
  const float* arm   = (const float*)d_in[7];
  const float* bm    = (const float*)d_in[8];
  const float* W2    = (const float*)d_in[9];
  const float* al2   = (const float*)d_in[10];
  const float* ar2   = (const float*)d_in[11];
  const float* b2    = (const float*)d_in[12];
  const float* gamma = (const float*)d_in[13];
  const float* beta  = (const float*)d_in[14];
  const int*   src   = (const int*)d_in[15];
  const int*   dst   = (const int*)d_in[16];
  (void)n_in; (void)out_size; (void)ws_size;

  const int N    = in_sizes[0] / 9;
  const int E    = in_sizes[15];
  const int Npad = ((N + 15) / 16) * 16;

  float* ws = (float*)d_ws;
  size_t off = 0;
  float* z    = ws + off; off += (size_t)N * 256;
  float* oacc = ws + off; off += (size_t)N * 256;
  float* hbuf = ws + off; off += (size_t)N * 64;
  float* el   = ws + off; off += (size_t)N * 4;
  float* er   = ws + off; off += (size_t)N * 4;
  float* mx   = ws + off; off += (size_t)N * 4;
  float* ss   = ws + off; off += (size_t)N * 4;
  float* ebuf = ws + off; off += (size_t)(E + N) * 4;
  float* stat = ws + off; off += 256;                 // colsum | colsq | mu | rsig
  _Float16* xh = (_Float16*)(ws + off); off += (size_t)Npad * 32;  // Npad*64 halves
  _Float16* wt = (_Float16*)(ws + off); off += 8192;               // 256*64 halves

  auto run_edges = [&](const float* zl, const float* al, const float* ar,
                       int outd, int ldz) {
    int NH = N * 4;
    attn_coeff<<<(NH + 255) / 256, 256, 0, stream>>>(zl, al, ar, el, er, N, outd, ldz);
    fill_kernel<<<(NH + 255) / 256, 256, 0, stream>>>(mx, NH, -1e30f);
    fill_kernel<<<(NH + 255) / 256, 256, 0, stream>>>(ss, NH, 0.f);
    int OC = N * 4 * outd;
    fill_kernel<<<(OC + 255) / 256, 256, 0, stream>>>(oacc, OC, 0.f);
    int EH = (E + N) * 4;
    edge_logits_max<<<(EH + 255) / 256, 256, 0, stream>>>(el, er, src, dst, E, N, ebuf, mx);
    edge_exp_sum<<<(EH + 255) / 256, 256, 0, stream>>>(ebuf, mx, ss, dst, E, N);
    long long tot = (long long)(E + N) * 4 * outd;
    unsigned blocks = (unsigned)((tot + 255) / 256);
    if (outd == 64)
      edge_scatter<64><<<blocks, 256, 0, stream>>>(ebuf, ss, zl, src, dst, E, N, oacc, ldz);
    else
      edge_scatter<2><<<blocks, 256, 0, stream>>>(ebuf, ss, zl, src, dst, E, N, oacc, ldz);
  };

  auto run_bn = [&](const float* b) {
    fill_kernel<<<1, 256, 0, stream>>>(stat, 128, 0.f);
    headsum_bn_stats<<<(N * 64 + 255) / 256, 256, 0, stream>>>(oacc, b, hbuf,
                                                               stat, stat + 64, N);
    finalize_stats<<<1, 64, 0, stream>>>(stat, stat + 64, stat + 128, stat + 192, N);
    bn_apply<<<(N * 64 + 255) / 256, 256, 0, stream>>>(hbuf, stat + 128, stat + 192,
                                                       gamma, beta, N);
  };

  // ---- Layer 1: [N,9] x [9,256], Kpad=32 ----
  {
    int nx = Npad * 32;
    cvt_x<<<(nx + 255) / 256, 256, 0, stream>>>(feats, xh, N, 9, 32, Npad);
    int nw = 256 * 32;
    cvt_wt<<<(nw + 255) / 256, 256, 0, stream>>>(W1, wt, 9, 256, 32, 256);
    dim3 g(Npad / 16, 256 / 64);
    gemm_wmma<4><<<g, 32, 0, stream>>>(xh, wt, z, N, 32, 256);
  }
  run_edges(z, al1, ar1, 64, 256);
  run_bn(b1);

  // ---- Layer 2 (middle): [N,64] x [64,256], Kpad=64 ----
  {
    int nx = Npad * 64;
    cvt_x<<<(nx + 255) / 256, 256, 0, stream>>>(hbuf, xh, N, 64, 64, Npad);
    int nw = 256 * 64;
    cvt_wt<<<(nw + 255) / 256, 256, 0, stream>>>(Wm, wt, 64, 256, 64, 256);
    dim3 g(Npad / 16, 256 / 64);
    gemm_wmma<4><<<g, 32, 0, stream>>>(xh, wt, z, N, 64, 256);
  }
  run_edges(z, alm, arm, 64, 256);
  run_bn(bm);

  // ---- Layer 3: [N,64] x [64,8] (cols padded to 16), Kpad=64 ----
  {
    int nx = Npad * 64;
    cvt_x<<<(nx + 255) / 256, 256, 0, stream>>>(hbuf, xh, N, 64, 64, Npad);
    int nw = 16 * 64;
    cvt_wt<<<(nw + 255) / 256, 256, 0, stream>>>(W2, wt, 64, 8, 64, 16);
    dim3 g(Npad / 16, 1);
    gemm_wmma<1><<<g, 32, 0, stream>>>(xh, wt, z, N, 64, 16);
  }
  run_edges(z, al2, ar2, 2, 16);
  headsum_final<<<(N * 2 + 255) / 256, 256, 0, stream>>>(oacc, b2, (float*)d_out, N);
}
